// ALayer_DR1_wh_light_v1_14637248545088
// MI455X (gfx1250) — compile-verified, matrix operand order checked by reading, not run on hardware
//
#include <hip/hip_runtime.h>
#include <math.h>

typedef _Float16 v8h  __attribute__((ext_vector_type(8)));
typedef _Float16 v16h __attribute__((ext_vector_type(16)));
typedef float    v8f  __attribute__((ext_vector_type(8)));
typedef unsigned int u32x4 __attribute__((ext_vector_type(4)));
typedef int          i32x4 __attribute__((ext_vector_type(4)));
typedef int          i32x8 __attribute__((ext_vector_type(8)));

#define BB   128
#define CC   1024
#define NSP  64      // 8*8 spatial
#define CK   9216    // 9*C
#define WIN  8192    // 8*C
#define WH   512     // WIN/16
#define CCH  32      // channels per main-loop chunk
#define KCH  288     // 9*CCH K-values per chunk (9 WMMA K-steps)

// ---------------------------------------------------------------- helpers
__device__ __forceinline__ v16h cat16(v8h lo, v8h hi) {
    return __builtin_shufflevector(lo, hi, 0,1,2,3,4,5,6,7,8,9,10,11,12,13,14,15);
}

// TDM: async copy of a 2D tile (CCH rows x 64 floats, row stride 64 floats)
// from global memory into LDS at byte offset lds_off. D# per ISA ch.8:
//  group0: count=1, lds_addr, global_addr[56:0], type=2
//  group1: data_size=4B(code 2), tensor_dim0=64, tensor_dim1=CCH,
//          tile_dim0=64, tile_dim1=CCH, tensor_dim0_stride=64
// 6-arg toolchain form: (g0, g1, g2, g3, g4, cpol); groups 2+ zero (<=2D tile).
__device__ __forceinline__ void tdm_load_x(unsigned lds_off, const float* gptr) {
    unsigned long long ga = (unsigned long long)(uintptr_t)gptr;
    u32x4 g0;
    g0[0] = 1u;                                   // count=1, user descriptor
    g0[1] = lds_off;                              // lds_addr (bytes)
    g0[2] = (unsigned)ga;                         // global_addr[31:0]
    g0[3] = ((unsigned)(ga >> 32) & 0x01FFFFFFu)  // global_addr[56:32]
          | (2u << 30);                           // type=2 ("image")
    i32x8 g1;
    g1[0] = (int)(2u << 16);                      // data_size=2 -> 4 bytes
    g1[1] = (int)(64u << 16);                     // tensor_dim0 = 64
    g1[2] = (int)((unsigned)CCH << 16);           // tensor_dim1 = CCH
    g1[3] = (int)(64u << 16);                     // tile_dim0 = 64
    g1[4] = (int)CCH;                             // tile_dim1 = CCH (tile_dim2=0)
    g1[5] = 64;                                   // tensor_dim0_stride = 64
    g1[6] = 0;
    g1[7] = 0;
    i32x4 gz4 = {0, 0, 0, 0};                     // groups 2/3 unused (<=2D)
    i32x8 gz8 = {0, 0, 0, 0, 0, 0, 0, 0};
    __builtin_amdgcn_tensor_load_to_lds(g0, g1, gz4, gz4, gz8, 0);
}

// ---------------------------------------------------------------- f32 -> f16 cast
__global__ __launch_bounds__(256)
void cvt_f16_kernel(const float* __restrict__ src, _Float16* __restrict__ dst, int n) {
    int i = blockIdx.x * 256 + threadIdx.x;
    if (i < n) dst[i] = (_Float16)src[i];
}

// ---------------------------------------------------------------- reductions over x
__global__ __launch_bounds__(64)
void reduce_kernel(const float* __restrict__ x,
                   _Float16* __restrict__ xw16,   // [B][C*8]  mean over h
                   _Float16* __restrict__ xh16,   // [B][C*8]  mean over w
                   float*    __restrict__ ymean)  // [B][C]
{
    int bc = blockIdx.x;                 // b*1024 + c
    int t  = threadIdx.x;
    __shared__ float s[64];
    s[t] = x[(size_t)bc * 64 + t];
    __syncthreads();
    int b = bc >> 10, c = bc & 1023;
    if (t < 8) {                         // xh[b, c*8 + h] = mean over w
        float sum = 0.f;
        for (int w = 0; w < 8; ++w) sum += s[t * 8 + w];
        xh16[(size_t)b * WIN + c * 8 + t] = (_Float16)(sum * 0.125f);
    } else if (t < 16) {                 // xw[b, c*8 + w] = mean over h
        int w = t - 8;
        float sum = 0.f;
        for (int h = 0; h < 8; ++h) sum += s[h * 8 + w];
        xw16[(size_t)b * WIN + c * 8 + w] = (_Float16)(sum * 0.125f);
    } else if (t == 16) {
        float sum = 0.f;
        for (int i = 0; i < 64; ++i) sum += s[i];
        ymean[(size_t)b * CC + c] = sum * (1.f / 64.f);
    }
}

// ---------------------------------------------------------------- SE MLP layer 1
__global__ __launch_bounds__(64)
void se1_kernel(const float* __restrict__ ymean, const float* __restrict__ f1,
                float* __restrict__ y2)          // [B][64], relu
{
    int b = blockIdx.x;
    __shared__ float s[CC];
    for (int i = threadIdx.x; i < CC; i += 64) s[i] = ymean[(size_t)b * CC + i];
    __syncthreads();
    int j = threadIdx.x;
    const float* fr = f1 + (size_t)j * CC;
    float acc = 0.f;
    for (int i = 0; i < CC; ++i) acc += fr[i] * s[i];
    y2[b * 64 + j] = acc > 0.f ? acc : 0.f;
}

// ---------------------------------------------------------------- SE MLP layer 2
__global__ __launch_bounds__(256)
void se2_kernel(const float* __restrict__ y2, const float* __restrict__ f2,
                float* __restrict__ ybig)        // [B][9216], sigmoid
{
    int b = blockIdx.y;
    int k = blockIdx.x * 256 + threadIdx.x;      // 36*256 == 9216 exactly
    __shared__ float s[64];
    if (threadIdx.x < 64) s[threadIdx.x] = y2[b * 64 + threadIdx.x];
    __syncthreads();
    const float* fr = f2 + (size_t)k * 64;
    float acc = 0.f;
    for (int j = 0; j < 64; ++j) acc += fr[j] * s[j];
    ybig[(size_t)b * CK + k] = 1.f / (1.f + __expf(-acc));
}

// ---------------------------------------------------------------- attention GEMM (WMMA)
// out[m,n] = relu( sum_k A16[m][k] * B16[n][k] ), M=128, N=512, K=8192
__global__ __launch_bounds__(32)
void attn_gemm_kernel(const _Float16* __restrict__ A16,   // [128][8192]
                      const _Float16* __restrict__ B16,   // [512][8192] (w1, column-major B)
                      float* __restrict__ outp)           // [128][512]
{
    const int n0   = blockIdx.x * 16;
    const int m0   = blockIdx.y * 16;
    const int lane = threadIdx.x;
    const int row  = lane & 15;
    const int ko   = (lane >> 4) << 3;

    const _Float16* pa = A16 + (size_t)(m0 + row) * WIN + ko;
    const _Float16* pb = B16 + (size_t)(n0 + row) * WIN + ko;

    v8f acc = {};
    for (int k0 = 0; k0 < WIN; k0 += 32) {
        v16h a  = cat16(*(const v8h*)(pa + k0), *(const v8h*)(pa + k0 + 16));
        v16h bf = cat16(*(const v8h*)(pb + k0), *(const v8h*)(pb + k0 + 16));
        acc = __builtin_amdgcn_wmma_f32_16x16x32_f16(false, a, false, bf,
                                                     (short)0, acc, false, false);
    }
    const int hi8 = (lane & 16) ? 8 : 0;
    #pragma unroll
    for (int v = 0; v < 8; ++v) {
        float r = acc[v];
        r = r > 0.f ? r : 0.f;
        outp[(size_t)(m0 + hi8 + v) * WH + n0 + row] = r;
    }
}

// ---------------------------------------------------------------- attention layer 2 + outer product
__global__ __launch_bounds__(64)
void attn2_kernel(const float* __restrict__ h1w, const float* __restrict__ h1h,
                  const float* __restrict__ w2, float* __restrict__ aspat)  // [B][64]
{
    int b = blockIdx.x;
    __shared__ float sw[WH], sh[WH], aw[8], ah[8];
    for (int i = threadIdx.x; i < WH; i += 64) {
        sw[i] = h1w[(size_t)b * WH + i];
        sh[i] = h1h[(size_t)b * WH + i];
    }
    __syncthreads();
    int t = threadIdx.x;
    if (t < 8) {
        const float* wr = w2 + (size_t)t * WH;
        float a = 0.f;
        for (int n = 0; n < WH; ++n) a += wr[n] * sw[n];
        aw[t] = 1.f / (1.f + __expf(-a));
    } else if (t < 16) {
        int i = t - 8;
        const float* wr = w2 + (size_t)i * WH;
        float a = 0.f;
        for (int n = 0; n < WH; ++n) a += wr[n] * sh[n];
        ah[i] = 1.f / (1.f + __expf(-a));
    }
    __syncthreads();
    aspat[b * 64 + t] = ah[t >> 3] * aw[t & 7];   // l = lh*8+lw
}

// ---------------------------------------------------------------- main implicit-GEMM conv (WMMA + TDM)
// out[b,o,l] = aspat[b,l] * sum_k wf16[o][k] * (uf[b,k,l]*ybig[b,k])
// block: (b, 128-row M tile), 8 waves x 16 rows, N=64 (4 tiles/wave).
// K chunked by 288 (32 channels); x slab staged via double-buffered TDM.
__global__ __launch_bounds__(256)
void main_gemm_kernel(const _Float16* __restrict__ wf16,   // [1024][9216]
                      const float* __restrict__ x,         // [128][1024][8][8]
                      const float* __restrict__ ybig,      // [128][9216]
                      const float* __restrict__ aspat,     // [128][64]
                      float* __restrict__ out)             // [128][1024][64]
{
    const int b    = blockIdx.x >> 3;
    const int mblk = blockIdx.x & 7;
    const int wave = threadIdx.x >> 5;
    const int lane = threadIdx.x & 31;
    const int row  = lane & 15;
    const int ko   = (lane >> 4) << 3;

    __shared__ __align__(16) _Float16 Bs[NSP][KCH];   // B^T tile, 36 KB
    __shared__ __align__(16) float    Xs[2][CCH][NSP];// x slabs, 2 x 8 KB (TDM dest)
    __shared__ float                  Ys[KCH];        // y chunk

    v8f acc[4] = {{}, {}, {}, {}};
    const int m0 = mblk * 128 + wave * 16;
    const _Float16* Arow = wf16 + (size_t)(m0 + row) * CK;
    const float* xb = x    + (size_t)b * CC * NSP;
    const float* yb = ybig + (size_t)b * CK;

    if (wave == 0)                                    // prefetch chunk 0
        tdm_load_x((unsigned)(uintptr_t)&Xs[0][0][0], xb);

    for (int ch = 0; ch < CC / CCH; ++ch) {           // 32 chunks
        const int buf = ch & 1;
        const int k0  = ch * KCH;
        // stage y chunk (cheap, regular loads)
        for (int e = threadIdx.x; e < KCH; e += 256) Ys[e] = yb[k0 + e];
        if (wave == 0)                                // TDM done for Xs[buf]
            __builtin_amdgcn_s_wait_tensorcnt((short)0);
        __syncthreads();                              // publish Xs[buf]/Ys; Bs free

        if (wave == 0 && ch + 1 < CC / CCH)           // overlap next slab with compute
            tdm_load_x((unsigned)(uintptr_t)&Xs[buf ^ 1][0][0],
                       xb + (size_t)(ch + 1) * CCH * NSP);

        // build modulated unfold tile from LDS (no global gather)
        for (int e = threadIdx.x; e < KCH * NSP; e += 256) {
            int l  = e & 63;
            int kk = e >> 6;                          // 0..287
            int cl = kk / 9;
            int t  = kk - cl * 9;
            int kh = t / 3;
            int kw = t - kh * 3;
            int ih = (l >> 3) + kh - 1;
            int iw = (l & 7)  + kw - 1;
            float v = 0.f;
            if ((unsigned)ih < 8u && (unsigned)iw < 8u)
                v = Xs[buf][cl][(ih << 3) + iw];
            Bs[l][kk] = (_Float16)(v * Ys[kk]);
        }
        __syncthreads();

        #pragma unroll
        for (int ks = 0; ks < 9; ++ks) {
            const _Float16* pa = Arow + k0 + ks * 32 + ko;
            v16h a = cat16(*(const v8h*)pa, *(const v8h*)(pa + 16));
            #pragma unroll
            for (int nt = 0; nt < 4; ++nt) {
                const _Float16* pb = &Bs[nt * 16 + row][ks * 32 + ko];
                v16h bf = cat16(*(const v8h*)pb, *(const v8h*)(pb + 16));
                acc[nt] = __builtin_amdgcn_wmma_f32_16x16x32_f16(
                    false, a, false, bf, (short)0, acc[nt], false, false);
            }
        }
        // loop-top barrier separates these WMMA reads from next Bs build
    }

    // store, scaling columns by spatial attention
    const int hi8 = (lane & 16) ? 8 : 0;
    float* ob = out + ((size_t)b * CC + m0 + hi8) * NSP;
    #pragma unroll
    for (int nt = 0; nt < 4; ++nt) {
        int n = nt * 16 + row;
        float sc = aspat[b * NSP + n];
        #pragma unroll
        for (int v = 0; v < 8; ++v)
            ob[(size_t)v * NSP + n] = acc[nt][v] * sc;
    }
}

// ---------------------------------------------------------------- host launcher
extern "C" void kernel_launch(void* const* d_in, const int* in_sizes, int n_in,
                              void* d_out, int out_size, void* d_ws, size_t ws_size,
                              hipStream_t stream) {
    const float* x      = (const float*)d_in[0];
    const float* weight = (const float*)d_in[1];
    const float* w1     = (const float*)d_in[2];
    const float* w2     = (const float*)d_in[3];
    const float* f1     = (const float*)d_in[4];
    const float* f2     = (const float*)d_in[5];
    float* out = (float*)d_out;

    char* ws = (char*)d_ws;
    size_t off = 0;
    auto alloc = [&](size_t bytes) -> void* {
        void* p = ws + off;
        off += (bytes + 255) & ~(size_t)255;
        return p;
    };
    _Float16* wf16  = (_Float16*)alloc((size_t)CC * CK * 2);   // 18.9 MB
    _Float16* w1f16 = (_Float16*)alloc((size_t)WH * WIN * 2);  //  8.4 MB
    _Float16* xw16  = (_Float16*)alloc((size_t)BB * WIN * 2);  //  2.1 MB
    _Float16* xh16  = (_Float16*)alloc((size_t)BB * WIN * 2);  //  2.1 MB
    float* ymean    = (float*)alloc((size_t)BB * CC * 4);
    float* y2       = (float*)alloc((size_t)BB * 64 * 4);
    float* ybig     = (float*)alloc((size_t)BB * CK * 4);      //  4.7 MB
    float* h1w      = (float*)alloc((size_t)BB * WH * 4);
    float* h1h      = (float*)alloc((size_t)BB * WH * 4);
    float* aspatb   = (float*)alloc((size_t)BB * NSP * 4);

    // precision casts for WMMA operands
    {
        int n = CC * CK;
        cvt_f16_kernel<<<(n + 255) / 256, 256, 0, stream>>>(weight, wf16, n);
        n = WH * WIN;
        cvt_f16_kernel<<<(n + 255) / 256, 256, 0, stream>>>(w1, w1f16, n);
    }
    // reductions over x
    reduce_kernel<<<BB * CC, 64, 0, stream>>>(x, xw16, xh16, ymean);
    // SE channel-attention path
    se1_kernel<<<BB, 64, 0, stream>>>(ymean, f1, y2);
    se2_kernel<<<dim3(CK / 256, BB), 256, 0, stream>>>(y2, f2, ybig);
    // factorized spatial-attention path (WMMA GEMMs)
    attn_gemm_kernel<<<dim3(WH / 16, BB / 16), 32, 0, stream>>>(xw16, w1f16, h1w);
    attn_gemm_kernel<<<dim3(WH / 16, BB / 16), 32, 0, stream>>>(xh16, w1f16, h1h);
    attn2_kernel<<<BB, 64, 0, stream>>>(h1w, h1h, w2, aspatb);
    // main implicit-GEMM conv with modulation
    main_gemm_kernel<<<BB * 8, 256, 0, stream>>>(wf16, x, ybig, aspatb, out);
}
